// GraphX_32710470927048
// MI455X (gfx1250) — compile-verified, matrix operand order checked by reading, not run on hardware
//
#include <hip/hip_runtime.h>
#include <cstdint>
#include <cstddef>

// ---------------- constants from the reference ----------------
#define HDIM   256
#define HEADS  4
#define DH     64
#define NNODES 128   // nodes per graph (126 feat + user + item)
#define FPG    126   // feature nodes per graph
#define NEG_SLOPE 0.2f

#define GATX_USE_ASYNC 1   // global_load_async_to_lds_b128 staging path

// ---------------- types ----------------
typedef __attribute__((ext_vector_type(16))) __bf16          v16bf;
typedef __attribute__((ext_vector_type(16))) unsigned short  v16u;
typedef __attribute__((ext_vector_type(8)))  float           v8f;
typedef __attribute__((ext_vector_type(8)))  float           f32x8;
typedef __attribute__((ext_vector_type(8)))  unsigned short  u16x8;

union ABFrag { v16u u; v16bf b; };

static __device__ __forceinline__ unsigned short f32_to_bf16(float f) {
  unsigned int u = __float_as_uint(f);
  u += 0x7fffu + ((u >> 16) & 1u);          // round-to-nearest-even
  return (unsigned short)(u >> 16);
}

// float atomic max via signed/unsigned int ordering trick (works on LDS too:
// lowers to ds_max_i32 / ds_min_u32 for shared addresses)
static __device__ __forceinline__ void atomicMaxF(float* addr, float val) {
  if (val >= 0.0f) atomicMax((int*)addr, __float_as_int(val));
  else             atomicMin((unsigned int*)addr, __float_as_uint(val));
}

// ---------------- CDNA5 async global->LDS helpers ---------------------------
static __device__ __forceinline__ void async_copy_b128(unsigned lds_off,
                                                       const void* gptr) {
  // VDST = LDS byte address (u32 VGPR), VADDR = 64-bit global address
  asm volatile("global_load_async_to_lds_b128 %0, %1, off"
               :: "v"(lds_off), "v"(gptr) : "memory");
}
static __device__ __forceinline__ void wait_async0() {
#if __has_builtin(__builtin_amdgcn_s_wait_asynccnt)
  __builtin_amdgcn_s_wait_asynccnt(0);
#else
  asm volatile("s_wait_asynccnt 0x0" ::: "memory");
#endif
}

// ---------------- weight convert + transpose: W[k][n] f32 -> Wt[n][k] bf16 --
__global__ void __launch_bounds__(256)
wcvt_kernel(const float* __restrict__ W, unsigned short* __restrict__ Wt) {
  int gid = blockIdx.x * 256 + threadIdx.x;    // 65536 elements
  int k = gid >> 8, n = gid & 255;
  Wt[n * 256 + k] = f32_to_bf16(W[gid]);       // W[gid] == W[k][n]
}

// ---------------- gather-GEMM: C[map(r)] = (tab[idx[r]] or tab[r]) @ W ------
// block = 128 threads (4 waves). blockIdx.x = M-tile (16 rows). Wave w owns
// columns [64w, 64w+64). 8 K-steps of v_wmma_f32_16x16x32_bf16, 4 N-subtiles.
// out node index = (r / rpg) * 128 + (r % rpg) + col_off
__global__ void __launch_bounds__(128)
gemm_bf16_wmma(const float* __restrict__ tab,
               const int*   __restrict__ idx,     // nullptr => identity gather
               const unsigned short* __restrict__ Wt,  // bf16 [N][K] = W^T
               float* __restrict__ C,
               int rows_per_graph, int col_off) {
  const int lane = threadIdx.x & 31;
  const int wave = threadIdx.x >> 5;
  const int half = lane >> 4;     // 0: K-low octet, 1: K-high octet
  const int l    = lane & 15;
  const int m0   = blockIdx.x * 16;
  const int n0   = wave * 64;

  const int arow = m0 + l;                       // A row handled by this lane
  const int srow = idx ? idx[arow] : arow;
  const float* ap = tab + (size_t)srow * HDIM;

  v8f acc[4];
  #pragma unroll
  for (int t = 0; t < 4; ++t)
    #pragma unroll
    for (int j = 0; j < 8; ++j) acc[t][j] = 0.0f;

  for (int k0 = 0; k0 < HDIM; k0 += 32) {
    const int ka = k0 + half * 8;
    // A fragment per ISA layout (lane<16: K=ka..ka+7 and ka+16..ka+23)
    f32x8 alo = *(const f32x8*)(ap + ka);
    f32x8 ahi = *(const f32x8*)(ap + ka + 16);
    ABFrag A;
    #pragma unroll
    for (int i = 0; i < 8; ++i) A.u[i]     = f32_to_bf16(alo[i]);
    #pragma unroll
    for (int i = 0; i < 8; ++i) A.u[8 + i] = f32_to_bf16(ahi[i]);

    #pragma unroll
    for (int t = 0; t < 4; ++t) {
      const unsigned short* bp = Wt + (size_t)(n0 + t * 16 + l) * HDIM + ka;
      u16x8 blo = *(const u16x8*)(bp);
      u16x8 bhi = *(const u16x8*)(bp + 16);
      ABFrag Bf;
      #pragma unroll
      for (int i = 0; i < 8; ++i) Bf.u[i]     = blo[i];
      #pragma unroll
      for (int i = 0; i < 8; ++i) Bf.u[8 + i] = bhi[i];
      acc[t] = __builtin_amdgcn_wmma_f32_16x16x32_bf16(
          false, A.b, false, Bf.b, (short)0, acc[t], false, false);
    }
  }

  // C layout: VGPR j, lane<16 -> M=j ; lane>=16 -> M=8+j ; N = lane%16
  #pragma unroll
  for (int j = 0; j < 8; ++j) {
    const int r    = m0 + j + 8 * half;
    const int node = (r / rows_per_graph) * NNODES + (r % rows_per_graph) + col_off;
    float* cp = C + (size_t)node * HDIM + n0 + l;
    #pragma unroll
    for (int t = 0; t < 4; ++t) cp[t * 16] = acc[t][j];
  }
}

// ---------------- fused per-graph GAT + readout ------------------------------
// One workgroup (256 threads = 8 waves) per graph. Entire per-graph state in
// LDS: h tile (128KB), message accumulator (128KB), attention scalars (~8KB).
// Total dynamic LDS = 270336 B (< 320KB per-WGP limit).
#define SMEM_FLOATS (2 * NNODES * HDIM + 4 * NNODES * HEADS)
#define SMEM_BYTES  (SMEM_FLOATS * 4)
#define EDGE_WAVES  8

__global__ void __launch_bounds__(256)
gat_fused_kernel(const float* __restrict__ h,        // [NT, 256]
                 const int* __restrict__ src, const int* __restrict__ dst,
                 const float* __restrict__ a_src, const float* __restrict__ a_dst,
                 const float* __restrict__ Wout, const float* __restrict__ bout,
                 float* __restrict__ logits, int epg) {
  extern __shared__ float smem[];
  float* h_l  = smem;                       // [128][256]
  float* accm = h_l  + NNODES * HDIM;       // [128][256]
  float* as_l = accm + NNODES * HDIM;       // [128][4]
  float* ad_l = as_l + NNODES * HEADS;      // [128][4]
  float* m_l  = ad_l + NNODES * HEADS;      // [128][4]
  float* dn_l = m_l  + NNODES * HEADS;      // [128][4]

  const int g    = blockIdx.x;
  const int tid  = threadIdx.x;
  const int lane = tid & 31;
  const int wv   = tid >> 5;
  const int base = g * NNODES;
  const float* hg = h + (size_t)base * HDIM;
  const int* es = src + g * epg;
  const int* ed = dst + g * epg;

  // ---- stage h tile into LDS (async DMA path) + zero accumulator ----------
#if GATX_USE_ASYNC
  {
    const unsigned lds0 = (unsigned)(uintptr_t)h_l;   // low 32b = LDS offset
    for (int i = tid * 16; i < NNODES * HDIM * 4; i += 256 * 16)
      async_copy_b128(lds0 + (unsigned)i, (const char*)hg + i);
  }
#else
  for (int i = tid * 4; i < NNODES * HDIM; i += 256 * 4)
    *(float4*)(h_l + i) = *(const float4*)(hg + i);
#endif
  {
    float4 z; z.x = z.y = z.z = z.w = 0.0f;
    for (int i = tid * 4; i < NNODES * HDIM; i += 256 * 4)
      *(float4*)(accm + i) = z;
  }
#if GATX_USE_ASYNC
  wait_async0();
#endif
  __syncthreads();

  // ---- attention logits per (node, head) from LDS; init max/denom ---------
  for (int w = tid; w < NNODES * HEADS; w += 256) {
    int node = w >> 2, head = w & 3;
    const float* hp = h_l + node * HDIM + head * DH;
    const float* sv = a_src + head * DH;
    const float* dv = a_dst + head * DH;
    float s_ = 0.0f, d_ = 0.0f;
    #pragma unroll 4
    for (int i = 0; i < DH; ++i) { float v = hp[i]; s_ += v * sv[i]; d_ += v * dv[i]; }
    as_l[w] = s_; ad_l[w] = d_;
    m_l[w] = -1e30f; dn_l[w] = 0.0f;
  }
  __syncthreads();

  // ---- pass 1: segment max over incoming edges (LDS atomics) --------------
  for (int j = tid; j < epg * HEADS; j += 256) {
    int e = j >> 2, head = j & 3;
    int ls = es[e] - base, ld = ed[e] - base;
    float v = as_l[ls * HEADS + head] + ad_l[ld * HEADS + head];
    v = v > 0.0f ? v : NEG_SLOPE * v;
    atomicMaxF(&m_l[ld * HEADS + head], v);
  }
  __syncthreads();
  for (int w = tid; w < NNODES * HEADS; w += 256)
    if (m_l[w] < -1e29f) m_l[w] = 0.0f;     // isfinite(m) ? m : 0
  __syncthreads();

  // ---- pass 2: softmax denominator ----------------------------------------
  for (int j = tid; j < epg * HEADS; j += 256) {
    int e = j >> 2, head = j & 3;
    int ls = es[e] - base, ld = ed[e] - base;
    float v = as_l[ls * HEADS + head] + ad_l[ld * HEADS + head];
    v = v > 0.0f ? v : NEG_SLOPE * v;
    atomicAdd(&dn_l[ld * HEADS + head], expf(v - m_l[ld * HEADS + head]));
  }
  __syncthreads();

  // ---- pass 3: weighted message scatter (wave per edge, ds_add_f32) -------
  // lane handles channels {lane + 32*i}: 32 distinct LDS banks per DS op;
  // channel lane+32*i lies in head (i>>1) uniformly across the wave.
  for (int e = wv; e < epg; e += EDGE_WAVES) {
    int ls = es[e] - base, ld = ed[e] - base;
    float a4[HEADS];
    #pragma unroll
    for (int hh = 0; hh < HEADS; ++hh) {
      float v = as_l[ls * HEADS + hh] + ad_l[ld * HEADS + hh];
      v = v > 0.0f ? v : NEG_SLOPE * v;
      a4[hh] = expf(v - m_l[ld * HEADS + hh]) / (dn_l[ld * HEADS + hh] + 1e-16f);
    }
    const float* hs = h_l + ls * HDIM;
    float* ap_ = accm + ld * HDIM;
    #pragma unroll
    for (int i = 0; i < 8; ++i) {
      int c = lane + 32 * i;
      atomicAdd(&ap_[c], hs[c] * a4[i >> 1]);
    }
  }
  __syncthreads();

  // ---- fused ELU + readout GEMV (wave per feature node) -------------------
  for (int node = wv; node < FPG; node += EDGE_WAVES) {
    float a = 0.0f;
    #pragma unroll
    for (int i = 0; i < 8; ++i) {
      int k = lane + 32 * i;
      float v = accm[node * HDIM + k];
      v = v > 0.0f ? v : (expf(v) - 1.0f);    // ELU
      a += v * Wout[k];
    }
    #pragma unroll
    for (int off = 16; off; off >>= 1) a += __shfl_xor(a, off, 32);
    if (lane == 0) logits[g * FPG + node] = a + bout[0];
  }
}

// ---------------- host-side orchestration -----------------------------------
extern "C" void kernel_launch(void* const* d_in, const int* in_sizes, int n_in,
                              void* d_out, int out_size, void* d_ws, size_t ws_size,
                              hipStream_t stream) {
  const int*   fid    = (const int*)d_in[0];
  const int*   userid = (const int*)d_in[1];
  const int*   itemid = (const int*)d_in[2];
  const int*   eidx   = (const int*)d_in[3];
  const float* ftab   = (const float*)d_in[4];
  const float* utab   = (const float*)d_in[5];
  const float* itab   = (const float*)d_in[6];
  const float* Wfeat  = (const float*)d_in[7];
  const float* Wuser  = (const float*)d_in[8];
  const float* Witem  = (const float*)d_in[9];
  const float* Wgat   = (const float*)d_in[10];
  const float* a_src  = (const float*)d_in[11];
  const float* a_dst  = (const float*)d_in[12];
  const float* Wout   = (const float*)d_in[13];
  const float* bout   = (const float*)d_in[14];
  float* logits = (float*)d_out;

  const int B  = in_sizes[1];         // 256 graphs
  const int Mf = in_sizes[0];         // B*F = 32256 feature nodes
  const int NT = B * NNODES;          // 32768 total nodes
  const int E  = in_sizes[3] / 2;     // 524288 edges
  const int epg = E / B;              // 2048 edges per graph
  const int* src = eidx;
  const int* dst = eidx + E;

  // workspace carve-up (~68 MB)
  char* ws = (char*)d_ws;
  size_t off = 0;
  float* x = (float*)(ws + off); off += (size_t)NT * HDIM * 4;
  float* h = (float*)(ws + off); off += (size_t)NT * HDIM * 4;
  unsigned short* WtF = (unsigned short*)(ws + off); off += 256 * 256 * 2;
  unsigned short* WtU = (unsigned short*)(ws + off); off += 256 * 256 * 2;
  unsigned short* WtI = (unsigned short*)(ws + off); off += 256 * 256 * 2;
  unsigned short* WtG = (unsigned short*)(ws + off); off += 256 * 256 * 2;

  // 1) weights -> bf16 transposed
  wcvt_kernel<<<256, 256, 0, stream>>>(Wfeat, WtF);
  wcvt_kernel<<<256, 256, 0, stream>>>(Wuser, WtU);
  wcvt_kernel<<<256, 256, 0, stream>>>(Witem, WtI);
  wcvt_kernel<<<256, 256, 0, stream>>>(Wgat,  WtG);

  // 2) embedding gather-GEMMs into x[B,128,256]
  gemm_bf16_wmma<<<Mf / 16, 128, 0, stream>>>(ftab, fid,    WtF, x, FPG,     0);
  gemm_bf16_wmma<<<B  / 16, 128, 0, stream>>>(utab, userid, WtU, x, 1,     FPG);
  gemm_bf16_wmma<<<B  / 16, 128, 0, stream>>>(itab, itemid, WtI, x, 1, FPG + 1);

  // 3) h = x @ W_gat (identity row map)
  gemm_bf16_wmma<<<NT / 16, 128, 0, stream>>>(x, nullptr, WtG, h, 1 << 30, 0);

  // 4) fused per-graph GAT softmax + message passing + ELU + readout
  gat_fused_kernel<<<B, 256, SMEM_BYTES, stream>>>(
      h, src, dst, a_src, a_dst, Wout, bout, logits, epg);
}
